// CharLSTM_53017076302085
// MI455X (gfx1250) — matrix operand order repaired and checked by reading, not compiled, not run on hardware
//
#include <hip/hip_runtime.h>
#include <hip/hip_bf16.h>

typedef __attribute__((ext_vector_type(16))) _Float16 v16h;
typedef __attribute__((ext_vector_type(8)))  _Float16 v8h;
typedef __attribute__((ext_vector_type(8)))  float    v8f;

#define T_STEPS 20
#define NSEQ    12800     // B*L = 256*50
#define EDIM    128
#define HDIM    128
#define G4      512       // 4*H

__device__ __forceinline__ float sigmoidf_fast(float v) {
    return 1.0f / (1.0f + __expf(-v));
}

union FragU { v16h v; v8h h[2]; };

__global__ __launch_bounds__(256)
void charlstm_fused_wmma(const int*   __restrict__ x,
                         const float* __restrict__ embed,
                         const float* __restrict__ Wih_f, const float* __restrict__ Whh_f,
                         const float* __restrict__ b_f,
                         const float* __restrict__ Wih_b, const float* __restrict__ Whh_b,
                         const float* __restrict__ b_b,
                         float* __restrict__ out)
{
    // 264 KB of the 320 KB WGP LDS: both weight matrices resident as f16,
    // plus the per-step 16x128 activation tiles for x-proj and h-proj.
    __shared__ _Float16 sWih[G4 * EDIM];   // gate-major, row = 128 halves (256 B)
    __shared__ _Float16 sWhh[G4 * HDIM];
    __shared__ _Float16 sX[16 * EDIM];     // embedded chars for current timestep
    __shared__ _Float16 sH[16 * HDIM];     // hidden state (f16) for current timestep
    __shared__ int      sLen[16];

    const int tid  = threadIdx.x;
    const int lane = tid & 31;
    const int wv   = tid >> 5;          // 8 waves
    const int dir  = blockIdx.y;        // 0 = forward, 1 = backward
    const int n0   = blockIdx.x * 16;   // first sequence of this block

    const float* Wih  = dir ? Wih_b : Wih_f;
    const float* Whh  = dir ? Whh_b : Whh_f;
    const float* bias = dir ? b_b   : b_f;

    // ---- stage weights into LDS as f16 (amortized over 20 timesteps) ----
    {
        const float4* src = (const float4*)Wih;
        for (int i = tid; i < G4 * EDIM / 4; i += 256) {
            float4 v = src[i];
            sWih[4*i+0] = (_Float16)v.x; sWih[4*i+1] = (_Float16)v.y;
            sWih[4*i+2] = (_Float16)v.z; sWih[4*i+3] = (_Float16)v.w;
        }
        src = (const float4*)Whh;
        for (int i = tid; i < G4 * HDIM / 4; i += 256) {
            float4 v = src[i];
            sWhh[4*i+0] = (_Float16)v.x; sWhh[4*i+1] = (_Float16)v.y;
            sWhh[4*i+2] = (_Float16)v.z; sWhh[4*i+3] = (_Float16)v.w;
        }
    }

    // ---- h0 = 0 ----
    for (int i = tid; i < 16 * HDIM; i += 256) sH[i] = (_Float16)0.0f;

    // ---- per-sequence valid lengths: len[n] = count(x[n,:] > 0) ----
    if (tid < 16) {
        const int* xr = x + (size_t)(n0 + tid) * T_STEPS;
        int cnt = 0;
        for (int t = 0; t < T_STEPS; ++t) cnt += (xr[t] > 0) ? 1 : 0;
        sLen[tid] = cnt;
    }

    // ---- gather embeddings for the first timestep ----
    {
        const int tt0 = dir ? (T_STEPS - 1) : 0;
        const int row = tid >> 4, seg = tid & 15;
        const int idx = x[(size_t)(n0 + row) * T_STEPS + tt0];
        const float* er = embed + (size_t)idx * EDIM + seg * 8;
        #pragma unroll
        for (int j = 0; j < 8; ++j) sX[row * EDIM + seg * 8 + j] = (_Float16)er[j];
    }

    __syncthreads();

    // ---- per-wave tile geometry (ISA 7.12.2 layouts) ----
    const int wc    = wv * 16;                // h-column slice of this wave
    const int colN  = lane & 15;              // D/B: N = lane & 15
    const int mbase = (lane & 16) ? 8 : 0;    // D: M = vgpr + (lane>=16 ? 8 : 0)
    const int arow  = lane & 15;              // A: M = lane & 15
    const int kA    = (lane & 16) ? 8 : 0;    // A: K chunk offset per half-wave
    const int kB    = (lane & 16) ? 8 : 0;    // B: K chunk offset per half-wave

    int lenv[8];
    #pragma unroll
    for (int r = 0; r < 8; ++r) lenv[r] = sLen[mbase + r];

    float bg[4];
    #pragma unroll
    for (int g = 0; g < 4; ++g) bg[g] = bias[g * HDIM + wc + colN];

    v8f c, hcur;
    #pragma unroll
    for (int r = 0; r < 8; ++r) { c[r] = 0.0f; hcur[r] = 0.0f; }

    for (int step = 0; step < T_STEPS; ++step) {
        const int tt = dir ? (T_STEPS - 1 - step) : step;

        // A fragments for x-proj and h-proj (16x32 f16 tiles, 4 k-steps)
        v16h ax[4], ah[4];
        #pragma unroll
        for (int k = 0; k < 4; ++k) {
            const int ko = k * 32 + kA;
            FragU ua, uh;
            ua.h[0] = *(const v8h*)(sX + arow * EDIM + ko);
            ua.h[1] = *(const v8h*)(sX + arow * EDIM + ko + 16);
            uh.h[0] = *(const v8h*)(sH + arow * HDIM + ko);
            uh.h[1] = *(const v8h*)(sH + arow * HDIM + ko + 16);
            ax[k] = ua.v; ah[k] = uh.v;
        }

        // 4 gate tiles x (4 + 4) k-steps = 32 v_wmma_f32_16x16x32_f16 per wave
        v8f gate[4];
        #pragma unroll
        for (int g = 0; g < 4; ++g) {
            v8f acc;
            #pragma unroll
            for (int r = 0; r < 8; ++r) acc[r] = bg[g];
            const int col = g * HDIM + wc + colN;
            #pragma unroll
            for (int k = 0; k < 4; ++k) {
                FragU ub;
                const _Float16* p = sWih + col * EDIM + k * 32 + kB;
                ub.h[0] = *(const v8h*)p;
                ub.h[1] = *(const v8h*)(p + 16);
                acc = __builtin_amdgcn_wmma_f32_16x16x32_f16(
                          false, ax[k], false, ub.v, (short)0, acc, false, false);
            }
            #pragma unroll
            for (int k = 0; k < 4; ++k) {
                FragU ub;
                const _Float16* p = sWhh + col * HDIM + k * 32 + kB;
                ub.h[0] = *(const v8h*)p;
                ub.h[1] = *(const v8h*)(p + 16);
                acc = __builtin_amdgcn_wmma_f32_16x16x32_f16(
                          false, ah[k], false, ub.v, (short)0, acc, false, false);
            }
            gate[g] = acc;
        }

        // LSTM cell update in f32; per-element valid-masking (row = mbase + r)
        #pragma unroll
        for (int r = 0; r < 8; ++r) {
            const float iv = sigmoidf_fast(gate[0][r]);
            const float fv = sigmoidf_fast(gate[1][r]);
            const float gv = tanhf(gate[2][r]);
            const float ov = sigmoidf_fast(gate[3][r]);
            const float cn = fv * c[r] + iv * gv;
            const float hn = ov * tanhf(cn);
            const bool valid = tt < lenv[r];
            c[r]    = valid ? cn : c[r];
            hcur[r] = valid ? hn : hcur[r];
        }

        __syncthreads();   // everyone done reading sX / sH

        // publish h (f16) for next step's A operand
        #pragma unroll
        for (int r = 0; r < 8; ++r)
            sH[(mbase + r) * HDIM + wc + colN] = (_Float16)hcur[r];

        // gather embeddings for the next timestep
        if (step + 1 < T_STEPS) {
            const int ttn = dir ? (T_STEPS - 2 - step) : (step + 1);
            const int row = tid >> 4, seg = tid & 15;
            const int idx = x[(size_t)(n0 + row) * T_STEPS + ttn];
            const float* er = embed + (size_t)idx * EDIM + seg * 8;
            #pragma unroll
            for (int j = 0; j < 8; ++j)
                sX[row * EDIM + seg * 8 + j] = (_Float16)er[j];
        }

        __syncthreads();   // sX / sH ready for next iteration
    }

    // ---- final hidden state -> out[(n), dir*128 + col] (f32, (N, 256)) ----
    #pragma unroll
    for (int r = 0; r < 8; ++r)
        out[(size_t)(n0 + mbase + r) * 256 + dir * 128 + wc + colN] = hcur[r];
}

extern "C" void kernel_launch(void* const* d_in, const int* in_sizes, int n_in,
                              void* d_out, int out_size, void* d_ws, size_t ws_size,
                              hipStream_t stream) {
    const int*   x     = (const int*)  d_in[0];
    const float* embed = (const float*)d_in[1];
    const float* Wih_f = (const float*)d_in[2];
    const float* Whh_f = (const float*)d_in[3];
    const float* b_f   = (const float*)d_in[4];
    const float* Wih_b = (const float*)d_in[5];
    const float* Whh_b = (const float*)d_in[6];
    const float* b_b   = (const float*)d_in[7];
    float* out = (float*)d_out;

    dim3 grid(NSEQ / 16, 2);   // 800 sequence tiles x 2 directions
    charlstm_fused_wmma<<<grid, 256, 0, stream>>>(
        x, embed, Wih_f, Whh_f, b_f, Wih_b, Whh_b, b_b, out);
}
